// AnalogAttentionSuper_62904091017982
// MI455X (gfx1250) — matrix operand order repaired and checked by reading, not compiled, not run on hardware
//
#include <hip/hip_runtime.h>

typedef __attribute__((ext_vector_type(16))) __bf16 bf16x16;
typedef __attribute__((ext_vector_type(8)))  __bf16 bf16x8;
typedef __attribute__((ext_vector_type(8)))  float  f32x8;

constexpr int CB = 64, CN = 197, CD = 640, CH = 10, CHD = 64, CNP = 224, CT = 32;
constexpr int CM = CB * CN;              // 12608 token rows
constexpr float CSCALE = 0.125f;         // HD^-0.5

__device__ inline f32x8 wmma_bf16(bf16x16 a, bf16x16 b, f32x8 c) {
  return __builtin_amdgcn_wmma_f32_16x16x32_bf16(false, a, false, b, (short)0, c,
                                                 false, false);
}

// A-fragment (16x32, 16-bit): lane holds row m=lane%16; K = 16*(lane/16) + j
// -> two contiguous 8-element loads per lane.
__device__ inline bf16x16 load_a16(const __bf16* p) {
  bf16x8 lo = *(const bf16x8*)(p);
  bf16x8 hi = *(const bf16x8*)(p + 8);
  return __builtin_shufflevector(lo, hi, 0, 1, 2, 3, 4, 5, 6, 7,
                                 8, 9, 10, 11, 12, 13, 14, 15);
}

// ---------------- f32 -> bf16 convert ----------------
__global__ void cvt_kernel(const float* __restrict__ s, __bf16* __restrict__ d, size_t n) {
  size_t i = (size_t)blockIdx.x * blockDim.x + threadIdx.x;
  size_t stride = (size_t)gridDim.x * blockDim.x;
  for (; i < n; i += stride) d[i] = (__bf16)s[i];
}

// tables: pad 30->32 rows; k-tables row-major (B-matrix), v-tables transposed (A-matrix)
__global__ void prep_tabs(const float* __restrict__ kv, const float* __restrict__ kh,
                          const float* __restrict__ vv, const float* __restrict__ vh,
                          __bf16* __restrict__ bkv, __bf16* __restrict__ bkh,
                          __bf16* __restrict__ tTv, __bf16* __restrict__ tTh) {
  for (int i = threadIdx.x; i < CT * CHD; i += blockDim.x) {
    int t = i / CHD, d = i - t * CHD;
    bool valid = t < 30;
    float a  = valid ? kv[t * CHD + d] : 0.f;
    float b_ = valid ? kh[t * CHD + d] : 0.f;
    float c  = valid ? vv[t * CHD + d] : 0.f;
    float e  = valid ? vh[t * CHD + d] : 0.f;
    bkv[i] = (__bf16)a;
    bkh[i] = (__bf16)b_;
    tTv[d * CT + t] = (__bf16)c;
    tTh[d * CT + t] = (__bf16)e;
  }
}

// zero the token pads of k (B,H,NP,HD) and vT (B,H,HD,NP)
__global__ void zeropad(__bf16* __restrict__ k, __bf16* __restrict__ vT) {
  const int per_bh = (CNP - CN) * CHD;
  const int total = CB * CH * per_bh;
  for (int i = blockIdx.x * blockDim.x + threadIdx.x; i < total;
       i += gridDim.x * blockDim.x) {
    int bh = i / per_bh;
    int rem = i - bh * per_bh;
    int t = CN + rem / CHD;
    int d = rem - (rem / CHD) * CHD;
    k[((size_t)bh * CNP + t) * CHD + d] = (__bf16)0.f;
    vT[((size_t)bh * CHD + d) * CNP + t] = (__bf16)0.f;
  }
}

// ---------------- NT GEMM: C[M,Nn] = A[M,K] * B[Nn,K]^T ----------------
// MODE 0: f32 output (+optional bias).  MODE 1: QKV epilogue scatter to bf16.
template <int MODE>
__global__ __launch_bounds__(256) void gemm_nt(
    const __bf16* __restrict__ A, const __bf16* __restrict__ Bm,
    float* __restrict__ C, const float* __restrict__ bias,
    int M, int Nn, int K,
    __bf16* __restrict__ qo, __bf16* __restrict__ ko, __bf16* __restrict__ vo) {
  const int lane = threadIdx.x & 31;
  const int wave = threadIdx.x >> 5;
  const int li = lane & 15, g = lane >> 4;
  const int wm = wave & 3, wn = wave >> 2;
  const int m0 = blockIdx.x * 128 + wm * 32;
  const int n0 = blockIdx.y * 128 + wn * 64;

  int rA0 = m0 + li;       if (rA0 > M - 1) rA0 = M - 1;
  int rA1 = m0 + 16 + li;  if (rA1 > M - 1) rA1 = M - 1;
  const __bf16* pa0 = A + (size_t)rA0 * K + 16 * g;
  const __bf16* pa1 = A + (size_t)rA1 * K + 16 * g;
  const __bf16* pb[4];
#pragma unroll
  for (int j = 0; j < 4; ++j)
    pb[j] = Bm + (size_t)(n0 + j * 16 + li) * K + 16 * g;

  f32x8 acc[2][4] = {};
  for (int k0 = 0; k0 < K; k0 += 32) {
    bf16x16 a0 = load_a16(pa0 + k0);
    bf16x16 a1 = load_a16(pa1 + k0);
#pragma unroll
    for (int j = 0; j < 4; ++j) {
      bf16x16 b = *(const bf16x16*)(pb[j] + k0);
      acc[0][j] = wmma_bf16(a0, b, acc[0][j]);
      acc[1][j] = wmma_bf16(a1, b, acc[1][j]);
    }
  }

#pragma unroll
  for (int i = 0; i < 2; ++i) {
#pragma unroll
    for (int r = 0; r < 8; ++r) {
      int row = m0 + i * 16 + r + 8 * g;
      if (row >= M) continue;
      if (MODE == 0) {
#pragma unroll
        for (int j = 0; j < 4; ++j) {
          int col = n0 + j * 16 + li;
          float v = acc[i][j][r] + (bias ? bias[col] : 0.f);
          C[(size_t)row * Nn + col] = v;
        }
      } else {
        int bb = row / CN, tt = row - bb * CN;
#pragma unroll
        for (int j = 0; j < 4; ++j) {
          int e = n0 + j * 16 + li;
          int s = e / CD;
          int rem = e - s * CD;
          int h = rem >> 6, d = rem & 63;
          float v = acc[i][j][r];
          size_t bh = (size_t)bb * CH + h;
          if (s == 0)      qo[(bh * CN + tt) * CHD + d] = (__bf16)(v * CSCALE);
          else if (s == 1) ko[(bh * CNP + tt) * CHD + d] = (__bf16)v;
          else             vo[(bh * CHD + d) * CNP + tt] = (__bf16)v;
        }
      }
    }
  }
}

// ---------------- P_v / P_h: q_scaled (rows x 64) * tab^T (64 x 32) ----------------
__global__ __launch_bounds__(128) void pgemm(
    const __bf16* __restrict__ q, const __bf16* __restrict__ tkv,
    const __bf16* __restrict__ tkh, float* __restrict__ Pv, float* __restrict__ Ph) {
  const int lane = threadIdx.x & 31;
  const int wave = threadIdx.x >> 5;
  const int li = lane & 15, g = lane >> 4;
  const size_t m0 = ((size_t)blockIdx.x * 4 + wave) * 16;  // rows, exact multiple
  const __bf16* pa = q + (m0 + li) * CHD + 16 * g;
  bf16x16 a0 = load_a16(pa);
  bf16x16 a1 = load_a16(pa + 32);
  for (int tb = 0; tb < 2; ++tb) {
    const __bf16* tab = tb ? tkh : tkv;
    float* P = tb ? Ph : Pv;
#pragma unroll
    for (int j = 0; j < 2; ++j) {
      const __bf16* pbj = tab + (size_t)(j * 16 + li) * CHD + 16 * g;
      f32x8 c = {};
      c = wmma_bf16(a0, *(const bf16x16*)(pbj), c);
      c = wmma_bf16(a1, *(const bf16x16*)(pbj + 32), c);
#pragma unroll
      for (int r = 0; r < 8; ++r)
        P[(m0 + r + 8 * g) * CT + j * 16 + li] = c[r];
    }
  }
}

// ---------------- fused attention: one wave per (b,h,16 q rows) ----------------
__global__ __launch_bounds__(128) void attn(
    const __bf16* __restrict__ qm, const __bf16* __restrict__ km,
    const __bf16* __restrict__ vTm,
    const float* __restrict__ Pv, const float* __restrict__ Ph,
    const __bf16* __restrict__ tTvv, const __bf16* __restrict__ tTvh,
    __bf16* __restrict__ outm) {
  __shared__ float smem[4 * 2048];
  const int lane = threadIdx.x & 31;
  const int wave = threadIdx.x >> 5;
  const int li = lane & 15, g = lane >> 4;
  float* Pl = smem + wave * 2048;  // [2][16 q][32 t] staged bias dot-products
  float* Wl = Pl + 1024;           // [2][16 q][32 t] probability bucket sums

  const int item = blockIdx.x * 4 + wave;       // 8320 items exactly
  const int bh = item / 13, qt = item - bh * 13;
  const int q0 = qt * 16;
  const int b = bh / CH, h = bh - b * CH;

  for (int i = lane; i < 512; i += 32) {
    int r = i >> 5, c = i & 31;
    int qg_ = q0 + r; if (qg_ > CN - 1) qg_ = CN - 1;
    size_t base = ((size_t)bh * CN + qg_) * CT + c;
    Pl[i] = Pv[base];
    Pl[512 + i] = Ph[base];
    Wl[i] = 0.f;
    Wl[512 + i] = 0.f;
  }
  __syncthreads();

  int qg = q0 + li; if (qg > CN - 1) qg = CN - 1;   // this lane's q column
  const bool qcls = (qg == 0);
  const int rq = qcls ? 0 : qg - 1;
  const int qvi = rq / 14, qhi = rq - (rq / 14) * 14;

  // loop-invariant Q^T B-fragments (col = q, K = d)
  const __bf16* qrow = qm + ((size_t)bh * CN + qg) * CHD + 16 * g;
  const bf16x16 qb0 = *(const bf16x16*)(qrow);
  const bf16x16 qb1 = *(const bf16x16*)(qrow + 32);

  auto relidx = [&](int key, int& tv, int& th) {
    if (qcls || key == 0) { tv = 0; th = 0; return; }
    int rk = key - 1;
    int dv = rk / 14 - qvi;
    int dh = rk - (rk / 14) * 14 - qhi;
    dv = dv < -14 ? -14 : (dv > 14 ? 14 : dv);
    dh = dh < -14 ? -14 : (dh > 14 ? 14 : dh);
    tv = dv + 15; th = dh + 15;
  };

  // S^T tile (16 keys x 16 q): A = K rows, B = Q^T
  auto score16 = [&](int kr0) -> f32x8 {
    const __bf16* kr = km + ((size_t)bh * CNP + kr0 + li) * CHD + 16 * g;
    f32x8 c = {};
    c = wmma_bf16(load_a16(kr), qb0, c);
    c = wmma_bf16(load_a16(kr + 32), qb1, c);
    return c;
  };

  // -------- pass 1: per-q max & sum (lane-local + one shfl) --------
  float mrow = -1e30f, lsum = 0.f;
  for (int kt = 0; kt < 7; ++kt) {
    const int k0 = kt * 32;
    f32x8 slo = score16(k0);
    f32x8 shi = score16(k0 + 16);
    float sv[16];
#pragma unroll
    for (int i = 0; i < 8; ++i) {
      int key = k0 + i + 8 * g;
      float s = -1e30f;
      if (key < CN) { int tv, th; relidx(key, tv, th);
        s = slo[i] + Pl[li * 32 + tv] + Pl[512 + li * 32 + th]; }
      sv[i] = s;
      key = k0 + 16 + i + 8 * g;
      s = -1e30f;
      if (key < CN) { int tv, th; relidx(key, tv, th);
        s = shi[i] + Pl[li * 32 + tv] + Pl[512 + li * 32 + th]; }
      sv[8 + i] = s;
    }
    float smax = -1e30f;
#pragma unroll
    for (int i = 0; i < 16; ++i) smax = fmaxf(smax, sv[i]);
    smax = fmaxf(smax, __shfl_xor(smax, 16, 32));
    float mn = fmaxf(mrow, smax);
    float ls = 0.f;
#pragma unroll
    for (int i = 0; i < 16; ++i) ls += __expf(sv[i] - mn);
    lsum = lsum * __expf(mrow - mn) + ls;
    mrow = mn;
  }
  lsum += __shfl_xor(lsum, 16, 32);
  const float linv = lsum > 0.f ? 1.f / lsum : 0.f;

  // -------- pass 2: normalized P, O^T += V^T * P^T, bucket sums --------
  f32x8 o[4] = {};
  for (int kt = 0; kt < 7; ++kt) {
    const int k0 = kt * 32;
    f32x8 slo = score16(k0);
    f32x8 shi = score16(k0 + 16);
    float plo[8], phi[8];
#pragma unroll
    for (int i = 0; i < 8; ++i) {
      int key = k0 + i + 8 * g;
      float p = 0.f;
      if (key < CN) {
        int tv, th; relidx(key, tv, th);
        float s = slo[i] + Pl[li * 32 + tv] + Pl[512 + li * 32 + th];
        p = __expf(s - mrow) * linv;
        atomicAdd(&Wl[li * 32 + tv], p);
        atomicAdd(&Wl[512 + li * 32 + th], p);
      }
      plo[i] = p;
      key = k0 + 16 + i + 8 * g;
      p = 0.f;
      if (key < CN) {
        int tv, th; relidx(key, tv, th);
        float s = shi[i] + Pl[li * 32 + tv] + Pl[512 + li * 32 + th];
        p = __expf(s - mrow) * linv;
        atomicAdd(&Wl[li * 32 + tv], p);
        atomicAdd(&Wl[512 + li * 32 + th], p);
      }
      phi[i] = p;
    }
    // assemble P^T B-fragment (32 keys x 16 q) via lane-pair exchange
    bf16x16 pB;
#pragma unroll
    for (int i = 0; i < 8; ++i) {
      float exlo = __shfl_xor(plo[i], 16, 32);
      float exhi = __shfl_xor(phi[i], 16, 32);
      pB[i]     = (__bf16)((g == 0) ? plo[i] : exhi);
      pB[8 + i] = (__bf16)((g == 0) ? exlo   : phi[i]);
    }
#pragma unroll
    for (int dt = 0; dt < 4; ++dt) {
      const __bf16* vr = vTm + ((size_t)bh * CHD + dt * 16 + li) * CNP + k0 + 16 * g;
      o[dt] = wmma_bf16(load_a16(vr), pB, o[dt]);
    }
  }

  __syncthreads();

  // value-side rel-pos: O^T += tabT_vv * W_v^T + tabT_vh * W_h^T
  bf16x16 bV, bH;
#pragma unroll
  for (int j = 0; j < 16; ++j) {
    bV[j] = (__bf16)Wl[li * 32 + 16 * g + j];
    bH[j] = (__bf16)Wl[512 + li * 32 + 16 * g + j];
  }
#pragma unroll
  for (int dt = 0; dt < 4; ++dt) {
    o[dt] = wmma_bf16(load_a16(tTvv + (dt * 16 + li) * CT + 16 * g), bV, o[dt]);
    o[dt] = wmma_bf16(load_a16(tTvh + (dt * 16 + li) * CT + 16 * g), bH, o[dt]);
  }

  const int qcol = q0 + li;
  if (qcol < CN) {
    __bf16* orow = outm + ((size_t)b * CN + qcol) * CD + h * CHD;
#pragma unroll
    for (int dt = 0; dt < 4; ++dt)
#pragma unroll
      for (int r = 0; r < 8; ++r)
        orow[dt * 16 + r + 8 * g] = (__bf16)o[dt][r];
  }
}

extern "C" void kernel_launch(void* const* d_in, const int* in_sizes, int n_in,
                              void* d_out, int out_size, void* d_ws, size_t ws_size,
                              hipStream_t stream) {
  const float* x     = (const float*)d_in[0];
  const float* wqkv  = (const float*)d_in[1];
  const float* wproj = (const float*)d_in[2];
  const float* bproj = (const float*)d_in[3];
  const float* tkv   = (const float*)d_in[4];
  const float* tkh   = (const float*)d_in[5];
  const float* tvv   = (const float*)d_in[6];
  const float* tvh   = (const float*)d_in[7];

  char* w = (char*)d_ws;
  size_t off = 0;
  auto alloc = [&](size_t bytes) {
    void* p = w + off;
    off += (bytes + 255) & ~(size_t)255;
    return p;
  };
  __bf16* x_bf   = (__bf16*)alloc((size_t)CM * CD * 2);
  __bf16* wq_bf  = (__bf16*)alloc((size_t)3 * CD * CD * 2);
  __bf16* wp_bf  = (__bf16*)alloc((size_t)CD * CD * 2);
  __bf16* tkv_bf = (__bf16*)alloc((size_t)CT * CHD * 2);
  __bf16* tkh_bf = (__bf16*)alloc((size_t)CT * CHD * 2);
  __bf16* tTvv   = (__bf16*)alloc((size_t)CHD * CT * 2);
  __bf16* tTvh   = (__bf16*)alloc((size_t)CHD * CT * 2);
  __bf16* q_bf   = (__bf16*)alloc((size_t)CB * CH * CN * CHD * 2);
  __bf16* k_bf   = (__bf16*)alloc((size_t)CB * CH * CNP * CHD * 2);
  __bf16* vT_bf  = (__bf16*)alloc((size_t)CB * CH * CNP * CHD * 2);
  float*  Pvb    = (float*)alloc((size_t)CB * CH * CN * CT * 4);
  float*  Phb    = (float*)alloc((size_t)CB * CH * CN * CT * 4);
  __bf16* out_bf = (__bf16*)alloc((size_t)CM * CD * 2);
  if (off > ws_size) return;  // insufficient workspace

  cvt_kernel<<<1024, 256, 0, stream>>>(x, x_bf, (size_t)CM * CD);
  cvt_kernel<<<256, 256, 0, stream>>>(wqkv, wq_bf, (size_t)3 * CD * CD);
  cvt_kernel<<<128, 256, 0, stream>>>(wproj, wp_bf, (size_t)CD * CD);
  prep_tabs<<<1, 256, 0, stream>>>(tkv, tkh, tvv, tvh, tkv_bf, tkh_bf, tTvv, tTvh);
  zeropad<<<4320, 256, 0, stream>>>(k_bf, vT_bf);

  // QKV projection + scatter (M=12608, N=1920, K=640)
  gemm_nt<1><<<dim3(99, 15), 256, 0, stream>>>(x_bf, wq_bf, nullptr, nullptr,
                                               CM, 3 * CD, CD, q_bf, k_bf, vT_bf);
  // P_v / P_h (126080 rows / 16 per wave / 4 waves per block)
  pgemm<<<1970, 128, 0, stream>>>(q_bf, tkv_bf, tkh_bf, Pvb, Phb);
  // fused attention (64*10*13 = 8320 wave-items)
  attn<<<2080, 128, 0, stream>>>(q_bf, k_bf, vT_bf, Pvb, Phb, tTvv, tTvh, out_bf);
  // output projection + bias (M=12608, N=640, K=640) -> f32 d_out
  gemm_nt<0><<<dim3(99, 5), 256, 0, stream>>>(out_bf, wp_bf, (float*)d_out, bproj,
                                              CM, CD, CD, nullptr, nullptr, nullptr);
}
